// FeatureAttention_17978733101171
// MI455X (gfx1250) — matrix operand order repaired and checked by reading, not compiled, not run on hardware
//
#include <hip/hip_runtime.h>
#include <hip/hip_bf16.h>
#include <cstdint>

typedef __attribute__((ext_vector_type(16))) _Float16 v16h;
typedef __attribute__((ext_vector_type(8)))  float    v8f;

// ---- problem constants (fixed by the reference) ----
constexpr int B_ = 16, T_ = 64, NF = 64, C_ = 512;
constexpr int TN = T_ * NF;                 // 4096 tokens per batch
constexpr long XB = (long)T_ * NF * C_;     // x batch stride (2M floats)
constexpr float SCALE = 0.044194173824159216f; // 1/sqrt(512)
constexpr float E_ = 2.718281828459045f;

// ---- workspace layout (float offsets) ----
constexpr size_t o_xsum   = 0;         // B*NF*C      = 524288
constexpr size_t o_weffT  = 524288;    // C*C         = 262144
constexpr size_t o_beff   = 786432;    // C
constexpr size_t o_wqb    = 786944;    // C
constexpr size_t o_cqb    = 787456;    // 64 (pad)
constexpr size_t o_qb     = 787520;    // B*NF
constexpr size_t o_Qp     = 788544;    // B*NF*C
constexpr size_t o_Vsum   = 1312832;   // B*NF*C
constexpr size_t o_VsumT  = 1837120;   // B*C*NF
constexpr size_t o_Vss    = 2361408;   // B*C
constexpr size_t o_rmax   = 2369600;   // B*NF
constexpr size_t o_rinv   = 2370624;   // B*NF
constexpr size_t o_dc     = 2371648;   // B*TN
constexpr size_t o_logits = 2437184;   // B*NF*TN = 4194304
constexpr size_t o_attnT  = 6631488;   // B*TN*NF = 4194304
// total ~43 MB

// =====================================================================
// Async helpers: CDNA5 global->LDS DMA (ASYNCcnt) + wait
// =====================================================================
__device__ __forceinline__ void async_cp32(uint32_t lds, const float* g) {
  // copies 32 bytes (8 floats) per lane; INST_OFFSET applies to both the
  // LDS and the global address, so one base pair covers both 16B chunks.
  asm volatile("global_load_async_to_lds_b128 %0, %1, off"
               :: "v"(lds), "v"(g) : "memory");
  asm volatile("global_load_async_to_lds_b128 %0, %1, off offset:16"
               :: "v"(lds), "v"(g) : "memory");
}
__device__ __forceinline__ void wait_async0() {
  asm volatile("s_wait_asynccnt 0x0" ::: "memory");
}

// assemble a v16h WMMA fragment from two 8-float LDS chunks (f32 -> f16)
__device__ __forceinline__ v16h frag_f32(const float* p0, const float* p1) {
  float4 a0 = *(const float4*)p0, a1 = *(const float4*)(p0 + 4);
  float4 b0 = *(const float4*)p1, b1 = *(const float4*)(p1 + 4);
  v16h r = { (_Float16)a0.x, (_Float16)a0.y, (_Float16)a0.z, (_Float16)a0.w,
             (_Float16)a1.x, (_Float16)a1.y, (_Float16)a1.z, (_Float16)a1.w,
             (_Float16)b0.x, (_Float16)b0.y, (_Float16)b0.z, (_Float16)b0.w,
             (_Float16)b1.x, (_Float16)b1.y, (_Float16)b1.z, (_Float16)b1.w };
  return r;
}

// =====================================================================
// Generic batched WMMA GEMM:  Cc[m,n] = sum_k A[m,k] * Bt[n,k]
// A: [M,K] row-major (lda=K), Bt: [N,K] row-major (B^T, ldb=K), C: [M,N].
// 64x64 tile / workgroup, 8 waves, each wave: two v_wmma_f32_16x16x32_f16
// per K-step. Double-buffered async global->LDS staging (f32 tiles),
// f32->f16 conversion at fragment assembly (co-executes with XDL WMMA).
// mode: 1 = +bias[col]*biasScale, then *outScale
//       2 = +bias[z*biasBatch+row], then *outScale   (logits + 1/sqrt(C))
//       3 = calib epilogue: omc*acc + cp*dc[z,row]*vss[z,col], NT stores
// =====================================================================
constexpr int LSF   = 36;        // f32 LDS row stride: 144B (16B aligned, low conflict)
constexpr int TILEF = 64 * LSF;  // floats per tile buffer

__global__ __launch_bounds__(256) void wmma_gemm(
    const float* __restrict__ A,  long aBatch,
    const float* __restrict__ Bt, long bBatch,
    float* __restrict__ Cc,       long cBatch,
    int K, int N,
    const float* __restrict__ bias, int biasBatch, float biasScale,
    int mode, float outScale,
    const float* __restrict__ dcp, const float* __restrict__ vss,
    const float* __restrict__ cpp)
{
  __shared__ __align__(16) float sAf[2][TILEF];
  __shared__ __align__(16) float sBf[2][TILEF];

  const int z  = blockIdx.z;
  const int mT = blockIdx.y * 64;
  const int nT = blockIdx.x * 64;
  const float* Ab = A  + (long)z * aBatch + (long)mT * K;
  const float* Bb = Bt + (long)z * bBatch + (long)nT * K;

  const int t     = threadIdx.x;
  const int ldRow = t >> 2;          // 0..63
  const int ldCol = (t & 3) * 8;     // 0,8,16,24
  const int w     = t >> 5;          // wave id (wave32), 0..7
  const int lane  = t & 31;
  const int rA    = (w & 3) * 16;    // wave's row block in tile
  const int cB    = (w >> 2) * 32;   // wave's col pair base
  const int lr    = lane & 15, lh = lane >> 4;

  // LDS byte addresses for this thread's staging slots (generic ptr low 32
  // bits == LDS offset per the flat-aperture mapping)
  const uint32_t aL = (uint32_t)(uintptr_t)&sAf[0][0];
  const uint32_t bL = (uint32_t)(uintptr_t)&sBf[0][0];
  const uint32_t stOff = (uint32_t)((ldRow * LSF + ldCol) * 4);

  auto issue = [&](int buf, int k0) {
    const float* ga = Ab + (long)ldRow * K + k0 + ldCol;
    const float* gb = Bb + (long)ldRow * K + k0 + ldCol;
    async_cp32(aL + (uint32_t)buf * (TILEF * 4) + stOff, ga);
    async_cp32(bL + (uint32_t)buf * (TILEF * 4) + stOff, gb);
  };

  v8f acc0 = {}; v8f acc1 = {};
  const int nsteps = K / 32;
  issue(0, 0);                       // prologue DMA
  for (int i = 0; i < nsteps; ++i) {
    const int buf = i & 1;
    wait_async0();                   // my async copies for `buf` are in LDS
    __syncthreads();                 // everyone's copies are in LDS
    if (i + 1 < nsteps) issue(buf ^ 1, (i + 1) * 32);  // overlap next DMA

    // A fragment (16x32): lane<16 row=lr K=0..15;
    // lane>=16 same row, K=16..31
    const float* ab = &sAf[buf][(rA + lr) * LSF];
    v16h fa = frag_f32(ab + lh * 16, ab + lh * 16 + 8);
    // B fragment (32x16): lane<16 col=lr K=0..15; lane>=16 K=16..31
    const float* b0p = &sBf[buf][(cB + lr) * LSF];
    v16h fb0 = frag_f32(b0p + lh * 16, b0p + lh * 16 + 8);
    const float* b1p = &sBf[buf][(cB + 16 + lr) * LSF];
    v16h fb1 = frag_f32(b1p + lh * 16, b1p + lh * 16 + 8);

    acc0 = __builtin_amdgcn_wmma_f32_16x16x32_f16(false, fa, false, fb0,
                                                  (short)0, acc0, false, false);
    acc1 = __builtin_amdgcn_wmma_f32_16x16x32_f16(false, fa, false, fb1,
                                                  (short)0, acc1, false, false);
  }

  // epilogue: acc VGPR r, lanes 0-15 -> M=r, lanes 16-31 -> M=8+r; N=lane&15
  const int col0 = nT + cB + lr;
  float* Cb = Cc + (long)z * cBatch;
  const float cpv = (mode == 3) ? cpp[0] : 0.f;
  const float omc = 1.f - cpv;
  for (int r = 0; r < 8; ++r) {
    const int row = mT + rA + lh * 8 + r;
    float v0 = acc0[r], v1 = acc1[r];
    if (mode == 1) {
      v0 += bias[col0] * biasScale; v1 += bias[col0 + 16] * biasScale;
      v0 *= outScale; v1 *= outScale;
      Cb[(long)row * N + col0]      = v0;
      Cb[(long)row * N + col0 + 16] = v1;
    } else if (mode == 2) {
      const float bb = bias[z * biasBatch + row];
      v0 = (v0 + bb) * outScale; v1 = (v1 + bb) * outScale;
      Cb[(long)row * N + col0]      = v0;
      Cb[(long)row * N + col0 + 16] = v1;
    } else { // mode == 3: final 128MB output -> non-temporal stores
      const float d = cpv * dcp[(long)z * TN + row];
      v0 = omc * v0 + d * vss[z * C_ + col0];
      v1 = omc * v1 + d * vss[z * C_ + col0 + 16];
      __builtin_nontemporal_store(v0, &Cb[(long)row * N + col0]);
      __builtin_nontemporal_store(v1, &Cb[(long)row * N + col0 + 16]);
    }
  }
}

// ---- xsum[b,n,c] = sum_t x[b,t,n,c] ----
__global__ void xsum_kernel(const float* __restrict__ x, float* __restrict__ xs) {
  const int idx = blockIdx.x * 256 + threadIdx.x;        // < B*NF*C
  const int b = idx >> 15; const int r = idx & 32767;
  const float* p = x + (long)b * XB + r;
  float s = 0.f;
  for (int tt = 0; tt < T_; ++tt) s += p[(long)tt * NF * C_];
  xs[idx] = s;
}

// ---- WeffT[d,c] = sum_e Wk[e,d]*Wq[e,c] ----
__global__ void weffT_kernel(const float* __restrict__ Wk,
                             const float* __restrict__ Wq,
                             float* __restrict__ WeffT) {
  const int idx = blockIdx.x * 256 + threadIdx.x;        // < C*C
  const int d = idx >> 9; const int c = idx & 511;
  float s = 0.f;
  for (int e = 0; e < C_; ++e) s += Wk[e * C_ + d] * Wq[e * C_ + c];
  WeffT[idx] = s;
}

// ---- beff[d]=bq@Wk ; wqb[c]=Wq^T@bk ; cqb=bq.bk ----
__global__ void biasprep_kernel(const float* __restrict__ Wk, const float* __restrict__ Wq,
                                const float* __restrict__ bq, const float* __restrict__ bk,
                                float* __restrict__ beff, float* __restrict__ wqb,
                                float* __restrict__ cqb) {
  const int g = blockIdx.x * 256 + threadIdx.x;          // < 512
  float s1 = 0.f, s2 = 0.f;
  for (int e = 0; e < C_; ++e) {
    s1 += bq[e] * Wk[e * C_ + g];
    s2 += bk[e] * Wq[e * C_ + g];
  }
  beff[g] = s1; wqb[g] = s2;
  if (g == 0) { float s3 = 0.f; for (int e = 0; e < C_; ++e) s3 += bq[e] * bk[e]; cqb[0] = s3; }
}

// ---- qb[b,n] = x_last[b,n,:].wqb + cqb ----
__global__ void qb_kernel(const float* __restrict__ x, const float* __restrict__ wqb,
                          const float* __restrict__ cqb, float* __restrict__ qb) {
  const int g = blockIdx.x * 256 + threadIdx.x;          // < B*NF
  const int b = g >> 6, n = g & 63;
  const float* xr = x + ((long)b * T_ + (T_ - 1)) * (long)NF * C_ + (long)n * C_;
  float s = cqb[0];
  for (int c = 0; c < C_; ++c) s += xr[c] * wqb[c];
  qb[g] = s;
}

// ---- Vss[b,c] = sum_n Vsum[b,n,c] ----
__global__ void vss_kernel(const float* __restrict__ Vsum, float* __restrict__ Vss) {
  const int g = blockIdx.x * 256 + threadIdx.x;          // < B*C
  const int b = g >> 9, c = g & 511;
  const float* p = Vsum + (long)b * NF * C_ + c;
  float s = 0.f;
  for (int n = 0; n < NF; ++n) s += p[n * C_];
  Vss[g] = s;
}

// ---- VsumT[b,c,n] = Vsum[b,n,c] (64x64 LDS tile transpose) ----
__global__ void vtr_kernel(const float* __restrict__ Vsum, float* __restrict__ VsumT) {
  __shared__ float tl[64][65];
  const int b = blockIdx.y; const int c0 = blockIdx.x * 64;
  const int t = threadIdx.x;
  const int n = t >> 2, cq = (t & 3) * 16;
  const float* p = Vsum + ((long)b * NF + n) * C_ + c0 + cq;
  for (int j = 0; j < 16; ++j) tl[n][cq + j] = p[j];
  __syncthreads();
  const int c = t >> 2, nq = (t & 3) * 16;
  float* q = VsumT + ((long)b * C_ + c0 + c) * NF + nq;
  for (int j = 0; j < 16; ++j) q[j] = tl[nq + j][c];
}

// ---- per-(b,n) softmax stats over 4096 logits ----
__global__ void softmax_stats(const float* __restrict__ logits,
                              float* __restrict__ rmax, float* __restrict__ rinv) {
  __shared__ float red[256];
  const float* L = logits + (long)blockIdx.x * TN;
  const int t = threadIdx.x;
  float m = -3.4e38f;
  for (int i = t; i < TN; i += 256) m = fmaxf(m, L[i]);
  red[t] = m; __syncthreads();
  for (int s = 128; s > 0; s >>= 1) { if (t < s) red[t] = fmaxf(red[t], red[t + s]); __syncthreads(); }
  m = red[0]; __syncthreads();
  float sum = 0.f;
  for (int i = t; i < TN; i += 256) sum += __expf(L[i] - m);
  red[t] = sum; __syncthreads();
  for (int s = 128; s > 0; s >>= 1) { if (t < s) red[t] += red[t + s]; __syncthreads(); }
  if (t == 0) { rmax[blockIdx.x] = m; rinv[blockIdx.x] = 1.f / red[0]; }
}

// ---- attnT[b,k,n] = softmax(logits)[b,n,k] via LDS transpose ----
__global__ void attnT_kernel(const float* __restrict__ logits,
                             const float* __restrict__ rmax, const float* __restrict__ rinv,
                             float* __restrict__ attnT) {
  __shared__ float tl[64][65];
  const int b = blockIdx.y; const int k0 = blockIdx.x * 64;
  const int t = threadIdx.x;
  const int n = t >> 2, kq = (t & 3) * 16;
  const int rid = b * NF + n;
  const float m = rmax[rid], inv = rinv[rid];
  const float* L = logits + (long)rid * TN + k0 + kq;
  for (int j = 0; j < 16; ++j) tl[n][kq + j] = __expf(L[j] - m) * inv;
  __syncthreads();
  const int k = t >> 2, nq = (t & 3) * 16;
  float* O = attnT + ((long)b * TN + k0 + k) * NF + nq;
  for (int j = 0; j < 16; ++j) O[j] = tl[nq + j][k];
}

// ---- dc[b,i] from a = mean_n attnT[b,i,:] ----
__global__ void dc_kernel(const float* __restrict__ attnT, const float* __restrict__ td,
                          const float* __restrict__ dec, const float* __restrict__ gfe,
                          const float* __restrict__ gmp, float* __restrict__ dc) {
  const int i = blockIdx.x * 256 + threadIdx.x;          // < B*TN
  const int kf = i & 63;
  const float* p = attnT + (long)i * NF;
  float s = 0.f;
  for (int n = 0; n < NF; ++n) s += p[n];
  const float a = s * (1.f / 64.f);
  const float tdc = fminf(td[i], 2.0f);
  const float dv = dec[kf] * logf(E_ + (1.f - a) * tdc);
  float d = tanhf(a / dv);
  if (tdc >= 2.0f) d = gmp[0] * gfe[kf];
  dc[i] = d;
}

// ---- dc_out[b,n,kf] = mean_t dc[b,t,kf] ----
__global__ void dcout_kernel(const float* __restrict__ dc, float* __restrict__ dco) {
  const int g = blockIdx.x * 256 + threadIdx.x;          // < B*NF
  const int b = g >> 6, kf = g & 63;
  float s = 0.f;
  for (int tt = 0; tt < T_; ++tt) s += dc[b * TN + tt * NF + kf];
  const float m = s * (1.f / 64.f);
  float* o = dco + (long)b * NF * NF + kf;
  for (int n = 0; n < NF; ++n) o[(long)n * NF] = m;
}

extern "C" void kernel_launch(void* const* d_in, const int* in_sizes, int n_in,
                              void* d_out, int out_size, void* d_ws, size_t ws_size,
                              hipStream_t stream) {
  (void)in_sizes; (void)n_in; (void)out_size; (void)ws_size;
  const float* x   = (const float*)d_in[0];
  const float* gfe = (const float*)d_in[1];
  const float* td  = (const float*)d_in[2];
  const float* Wq  = (const float*)d_in[3];
  const float* bq  = (const float*)d_in[4];
  const float* Wk  = (const float*)d_in[5];
  const float* bk  = (const float*)d_in[6];
  const float* Wv  = (const float*)d_in[7];
  const float* bv  = (const float*)d_in[8];
  const float* cpp = (const float*)d_in[9];
  const float* dec = (const float*)d_in[10];
  const float* gmp = (const float*)d_in[11];
  float* out = (float*)d_out;
  float* ws  = (float*)d_ws;

  // 1) time-sum of x (removes the big V projection)
  xsum_kernel<<<(B_ * NF * C_) / 256, 256, 0, stream>>>(x, ws + o_xsum);
  // 2) folded weights for K-free logits
  weffT_kernel<<<(C_ * C_) / 256, 256, 0, stream>>>(Wk, Wq, ws + o_weffT);
  biasprep_kernel<<<2, 256, 0, stream>>>(Wk, Wq, bq, bk, ws + o_beff, ws + o_wqb, ws + o_cqb);
  // 3) Q' = x_last @ WeffT^T + beff      [per-batch M=64, N=512, K=512]
  wmma_gemm<<<dim3(C_ / 64, 1, B_), 256, 0, stream>>>(
      x + (long)(T_ - 1) * NF * C_, XB, ws + o_weffT, 0L, ws + o_Qp, (long)NF * C_,
      C_, C_, ws + o_beff, 0, 1.f, 1, 1.f, nullptr, nullptr, nullptr);
  // 4) qb = x_last.wqb + cqb
  qb_kernel<<<(B_ * NF) / 256, 256, 0, stream>>>(x, ws + o_wqb, ws + o_cqb, ws + o_qb);
  // 5) Vsum = xsum @ Wv^T + T*bv         [M=1024, N=512, K=512]
  wmma_gemm<<<dim3(C_ / 64, (B_ * NF) / 64, 1), 256, 0, stream>>>(
      ws + o_xsum, 0L, Wv, 0L, ws + o_Vsum, 0L,
      C_, C_, bv, 0, (float)T_, 1, 1.f, nullptr, nullptr, nullptr);
  // 6) column sums + transpose of Vsum
  vss_kernel<<<(B_ * C_) / 256, 256, 0, stream>>>(ws + o_Vsum, ws + o_Vss);
  vtr_kernel<<<dim3(C_ / 64, B_), 256, 0, stream>>>(ws + o_Vsum, ws + o_VsumT);
  // 7) logits = (Q' @ x^T + qb) * 1/sqrt(C)   [per-batch M=64, N=4096, K=512]
  wmma_gemm<<<dim3(TN / 64, 1, B_), 256, 0, stream>>>(
      ws + o_Qp, (long)NF * C_, x, XB, ws + o_logits, (long)NF * TN,
      C_, TN, ws + o_qb, NF, 1.f, 2, SCALE, nullptr, nullptr, nullptr);
  // 8) softmax (stats, then exp + transpose to attnT)
  softmax_stats<<<B_ * NF, 256, 0, stream>>>(ws + o_logits, ws + o_rmax, ws + o_rinv);
  attnT_kernel<<<dim3(TN / 64, B_), 256, 0, stream>>>(
      ws + o_logits, ws + o_rmax, ws + o_rinv, ws + o_attnT);
  // 9) confidence values
  dc_kernel<<<(B_ * TN) / 256, 256, 0, stream>>>(ws + o_attnT, td, dec, gfe, gmp, ws + o_dc);
  dcout_kernel<<<(B_ * NF) / 256, 256, 0, stream>>>(ws + o_dc, out + (size_t)B_ * TN * C_);
  // 10) out = (1-cp)*(attnT @ VsumT^T) + cp*dc*Vss   [per-batch M=4096, N=512, K=64]
  wmma_gemm<<<dim3(C_ / 64, TN / 64, B_), 256, 0, stream>>>(
      ws + o_attnT, (long)TN * NF, ws + o_VsumT, (long)C_ * NF, out, (long)TN * C_,
      NF, C_, nullptr, 0, 0.f, 3, 1.f, ws + o_dc, ws + o_Vss, cpp);
}
